// MILAttentionPool_74088185856803
// MI455X (gfx1250) — matrix operand order repaired and checked by reading, not compiled, not run on hardware
//
#include <hip/hip_runtime.h>
#include <hip/hip_bf16.h>

// Problem constants (from reference): B=32, L=2048, D=512, H=256, K=4
#define BAGS 32
#define LBAG 2048
#define DIM  512
#define HID  256
#define NHEAD 4
#define NROWS (BAGS * LBAG)           // 65536 instances

typedef __attribute__((ext_vector_type(16))) _Float16 v16h;
typedef __attribute__((ext_vector_type(8)))  _Float16 v8h;
typedef __attribute__((ext_vector_type(4)))  _Float16 v4h;
typedef __attribute__((ext_vector_type(8)))  float    v8f;

// Branchless fast gate helpers (avoid OCML tanh's saveexec slow path).
__device__ __forceinline__ float fast_sigmoid(float t) {
    return 1.0f / (1.0f + __expf(-t));
}
__device__ __forceinline__ float fast_tanh(float t) {
    // tanh(t) = 1 - 2/(1 + exp(2t)); exact at +-inf of the exp range.
    return 1.0f - 2.0f / (1.0f + __expf(2.0f * t));
}

// ---------------------------------------------------------------------------
// Kernel 0: W1/W3 [D,H] fp32 -> transposed f16 [H,D] so B-fragments are
// contiguous 32B per-lane loads.
// ---------------------------------------------------------------------------
__global__ __launch_bounds__(128) void mil_convert_weights(
    const float* __restrict__ W1, const float* __restrict__ W3,
    _Float16* __restrict__ w1t, _Float16* __restrict__ w3t)
{
    const int n = blockIdx.x;                       // H index
    const float*   src = blockIdx.y ? W3  : W1;
    _Float16*      dst = blockIdx.y ? w3t : w1t;
    for (int k = threadIdx.x; k < DIM; k += 128)
        dst[n * DIM + k] = (_Float16)src[k * HID + n];
}

// ---------------------------------------------------------------------------
// Kernel 1: fused gated-attention MLP logits:
//   A[r, 0..3] = (tanh(x@W1+b1) * sigmoid(x@W3+b3)) @ W2 + b2
// 64 rows per block (4 waves x 16-row fragments), WMMA f32_16x16x32_f16.
// x tile staged in LDS as f16 with 16B XOR swizzle (conflict-free reads).
// ---------------------------------------------------------------------------
__device__ __forceinline__ int sx_off(int row, int kh) {
    // 16-byte-chunk XOR swizzle within a 512-half row
    return row * DIM + ((((kh >> 3) ^ (row & 15)) << 3) + (kh & 7));
}

__global__ __launch_bounds__(128) void mil_logits_kernel(
    const float* __restrict__ x,
    const _Float16* __restrict__ w1t, const _Float16* __restrict__ w3t,
    const float* __restrict__ b1, const float* __restrict__ b3,
    const float* __restrict__ W2, const float* __restrict__ b2,
    float* __restrict__ Alog)
{
    __shared__ _Float16 sx[64 * DIM];               // exactly 64 KB

    const int tid  = threadIdx.x;
    const int row0 = blockIdx.x * 64;

    // ---- stage 64x512 fp32 tile -> f16 LDS (swizzled) ----
    const float4* xg = (const float4*)(x + (size_t)row0 * DIM);
    #pragma unroll 4
    for (int i = 0; i < 64; ++i) {
        const int e4 = i * 128 + tid;               // float4 index in tile
        const int r  = e4 >> 7;                     // 128 float4 per row
        const int c4 = e4 & 127;
        const float4 v = xg[e4];
        v4h h; h[0] = (_Float16)v.x; h[1] = (_Float16)v.y;
               h[2] = (_Float16)v.z; h[3] = (_Float16)v.w;
        *(v4h*)&sx[sx_off(r, c4 * 4)] = h;
    }
    __syncthreads();

    const int lane  = tid & 31;
    const int wave  = tid >> 5;
    const int mrow  = lane & 15;                    // fragment row / N column
    const int mbase = wave * 16;
    const int hi    = (lane >= 16) ? 1 : 0;
    const int arow  = mbase + mrow;
    const int a_sel = hi ? 16 : 0;                  // A frag: {0-15} vs {16-31}
    const int b_sel = hi ? 16 : 0;                  // B frag: {0-15} vs {16-31}

    float pk[8][4];                                 // per-lane partials of gated@W2
    #pragma unroll
    for (int v = 0; v < 8; ++v)
        #pragma unroll
        for (int j = 0; j < 4; ++j) pk[v][j] = 0.f;

    for (int nt = 0; nt < 16; ++nt) {               // H tiles of 16
        const int n0   = nt * 16;
        const int ncol = n0 + mrow;                 // this lane's H column (C layout: N=lane%16)
        const _Float16* bp1 = w1t + (size_t)ncol * DIM + b_sel;
        const _Float16* bp3 = w3t + (size_t)ncol * DIM + b_sel;

        // Prefetch next H tile's weight rows into WGP$ (global_prefetch_b8).
        if (nt < 15) {
            __builtin_prefetch(bp1 + 16 * DIM, 0, 0);
            __builtin_prefetch(bp3 + 16 * DIM, 0, 0);
        }

        v8f c1 = {}; v8f c3 = {};
        #pragma unroll 8
        for (int kt = 0; kt < 16; ++kt) {           // D in steps of 32
            const int k0 = kt * 32;
            v8h alo = *(const v8h*)&sx[sx_off(arow, k0 + a_sel)];
            v8h ahi = *(const v8h*)&sx[sx_off(arow, k0 + a_sel + 8)];
            v16h af = __builtin_shufflevector(alo, ahi,
                        0,1,2,3,4,5,6,7,8,9,10,11,12,13,14,15);
            v16h bf1 = *(const v16h*)(bp1 + k0);
            v16h bf3 = *(const v16h*)(bp3 + k0);
            c1 = __builtin_amdgcn_wmma_f32_16x16x32_f16(
                    false, af, false, bf1, (short)0, c1, false, false);
            c3 = __builtin_amdgcn_wmma_f32_16x16x32_f16(
                    false, af, false, bf3, (short)0, c3, false, false);
        }

        // ---- fused epilogue: bias + tanh*sigmoid, project onto W2 heads ----
        const float bb1 = b1[ncol];
        const float bb3 = b3[ncol];
        const float4 w2v = *(const float4*)(W2 + ncol * 4);
        #pragma unroll
        for (int v = 0; v < 8; ++v) {
            const float g = fast_tanh(c1[v] + bb1) * fast_sigmoid(c3[v] + bb3);
            pk[v][0] += g * w2v.x; pk[v][1] += g * w2v.y;
            pk[v][2] += g * w2v.z; pk[v][3] += g * w2v.w;
        }
    }

    // ---- reduce partials across the 16 lanes of each half-wave ----
    #pragma unroll
    for (int v = 0; v < 8; ++v)
        #pragma unroll
        for (int j = 0; j < 4; ++j) {
            float s = pk[v][j];
            s += __shfl_xor(s, 8, 16);
            s += __shfl_xor(s, 4, 16);
            s += __shfl_xor(s, 2, 16);
            s += __shfl_xor(s, 1, 16);
            pk[v][j] = s;
        }

    if (mrow == 0) {
        const int m0 = row0 + mbase + (hi ? 8 : 0); // C layout: VGPR v -> M=v (+8 hi)
        #pragma unroll
        for (int v = 0; v < 8; ++v)
            #pragma unroll
            for (int j = 0; j < 4; ++j)
                Alog[(size_t)(m0 + v) * NHEAD + j] = pk[v][j] + b2[j];
    }
}

// ---------------------------------------------------------------------------
// Kernel 2: per-(bag,head) softmax over L=2048.  Alog [N,4] -> Asm [B,4,L]
// ---------------------------------------------------------------------------
__global__ __launch_bounds__(256) void mil_softmax_kernel(
    const float* __restrict__ Alog, float* __restrict__ Asm)
{
    const int b = blockIdx.x >> 2;
    const int k = blockIdx.x & 3;
    const int tid = threadIdx.x;
    __shared__ float red[256];

    const float* base = Alog + (size_t)b * LBAG * NHEAD + k;
    float vals[8];
    float lmax = -3.4e38f;
    #pragma unroll
    for (int i = 0; i < 8; ++i) {
        vals[i] = base[(size_t)(i * 256 + tid) * NHEAD];
        lmax = fmaxf(lmax, vals[i]);
    }
    red[tid] = lmax; __syncthreads();
    for (int s = 128; s > 0; s >>= 1) {
        if (tid < s) red[tid] = fmaxf(red[tid], red[tid + s]);
        __syncthreads();
    }
    const float m = red[0];
    __syncthreads();

    float lsum = 0.f;
    #pragma unroll
    for (int i = 0; i < 8; ++i) { vals[i] = __expf(vals[i] - m); lsum += vals[i]; }
    red[tid] = lsum; __syncthreads();
    for (int s = 128; s > 0; s >>= 1) {
        if (tid < s) red[tid] += red[tid + s];
        __syncthreads();
    }
    const float inv = 1.0f / red[0];

    float* out = Asm + ((size_t)b * NHEAD + k) * LBAG;
    #pragma unroll
    for (int i = 0; i < 8; ++i) out[i * 256 + tid] = vals[i] * inv;
}

// ---------------------------------------------------------------------------
// Kernel 3: attention-weighted pooling, L split 16 ways for occupancy.
// partial[s][b][k][d] = sum over 128 instances.  x read exactly once.
// ---------------------------------------------------------------------------
__global__ __launch_bounds__(512) void mil_wsum_partial_kernel(
    const float* __restrict__ x, const float* __restrict__ Asm,
    float* __restrict__ part)
{
    const int b = blockIdx.x;
    const int s = blockIdx.y;
    const int d = threadIdx.x;

    const float* xb = x + ((size_t)b * LBAG + s * 128) * DIM + d;
    const float* ab = Asm + (size_t)b * NHEAD * LBAG + s * 128;

    float a0 = 0.f, a1 = 0.f, a2 = 0.f, a3 = 0.f;
    #pragma unroll 4
    for (int l = 0; l < 128; ++l) {
        const float xv = xb[(size_t)l * DIM];
        a0 += xv * ab[l];
        a1 += xv * ab[LBAG + l];
        a2 += xv * ab[2 * LBAG + l];
        a3 += xv * ab[3 * LBAG + l];
    }
    float* p = part + (((size_t)s * BAGS + b) * NHEAD) * DIM + d;
    p[0 * DIM] = a0; p[1 * DIM] = a1; p[2 * DIM] = a2; p[3 * DIM] = a3;
}

// ---------------------------------------------------------------------------
// Kernel 4: reduce the 16 L-splits -> out[b, k*512 + d]
// ---------------------------------------------------------------------------
__global__ __launch_bounds__(256) void mil_wsum_reduce_kernel(
    const float* __restrict__ part, float* __restrict__ out)
{
    const int i = blockIdx.x * 256 + threadIdx.x;   // [0, 65536)
    float s = 0.f;
    #pragma unroll
    for (int j = 0; j < 16; ++j) s += part[(size_t)j * (BAGS * NHEAD * DIM) + i];
    out[i] = s;
}

// ---------------------------------------------------------------------------
extern "C" void kernel_launch(void* const* d_in, const int* in_sizes, int n_in,
                              void* d_out, int out_size, void* d_ws, size_t ws_size,
                              hipStream_t stream)
{
    (void)in_sizes; (void)n_in; (void)out_size; (void)ws_size;

    const float* x  = (const float*)d_in[0];
    const float* W1 = (const float*)d_in[1];
    const float* b1 = (const float*)d_in[2];
    const float* W3 = (const float*)d_in[3];
    const float* b3 = (const float*)d_in[4];
    const float* W2 = (const float*)d_in[5];
    const float* b2 = (const float*)d_in[6];
    // d_in[7] = bag_lengths (equal-length fast path; unused)

    char* ws = (char*)d_ws;
    _Float16* w1t  = (_Float16*)(ws + 0);                    // 256 KB
    _Float16* w3t  = (_Float16*)(ws + 262144);               // 256 KB
    float*    Alog = (float*)   (ws + 524288);               // 1 MB  [N,4]
    float*    Asm  = (float*)   (ws + 1572864);              // 1 MB  [B,4,L]
    float*    part = (float*)   (ws + 2621440);              // 4 MB  [16,B,4,D]
    float*    out  = (float*)d_out;

    mil_convert_weights<<<dim3(HID, 2), 128, 0, stream>>>(W1, W3, w1t, w3t);
    mil_logits_kernel<<<NROWS / 64, 128, 0, stream>>>(
        x, w1t, w3t, b1, b3, W2, b2, Alog);
    mil_softmax_kernel<<<BAGS * NHEAD, 256, 0, stream>>>(Alog, Asm);
    mil_wsum_partial_kernel<<<dim3(BAGS, 16), 512, 0, stream>>>(x, Asm, part);
    mil_wsum_reduce_kernel<<<(BAGS * NHEAD * DIM) / 256, 256, 0, stream>>>(part, out);
}
